// SP_52063593562599
// MI455X (gfx1250) — matrix operand order repaired and checked by reading, not compiled, not run on hardware
//
#include <hip/hip_runtime.h>
#include <stdint.h>

// Shapes fixed by the reference.
#define BATCH 16384
#define DIN   512
#define DOUT  4096

typedef __attribute__((ext_vector_type(16))) __bf16 v16bf;
typedef __attribute__((ext_vector_type(8)))  __bf16 v8bf;
typedef __attribute__((ext_vector_type(8)))  float  v8f;
typedef int v4i __attribute__((vector_size(16)));   // matches builtin param type

#if defined(__has_builtin)
#if __has_builtin(__builtin_amdgcn_global_load_async_to_lds_b128)
#define USE_ASYNC_LDS 1
#endif
#endif

// round-to-nearest-even f32 -> bf16
__device__ __forceinline__ unsigned short f2bf(float f) {
  union { float f; unsigned u; } x; x.f = f;
  unsigned r = x.u + 0x7FFFu + ((x.u >> 16) & 1u);
  return (unsigned short)(r >> 16);
}

__device__ __forceinline__ void wait_async_all() {
#ifdef USE_ASYNC_LDS
#if __has_builtin(__builtin_amdgcn_s_wait_asynccnt)
  __builtin_amdgcn_s_wait_asynccnt(0);
#else
  asm volatile("s_wait_asynccnt 0" ::: "memory");
#endif
#endif
}

__device__ __forceinline__ void copy16_to_lds(const unsigned short* g,
                                              unsigned short* l) {
#ifdef USE_ASYNC_LDS
  __builtin_amdgcn_global_load_async_to_lds_b128(
      (__attribute__((address_space(1))) v4i*)g,
      (__attribute__((address_space(3))) v4i*)l, 0, 0);
#else
  *(uint4*)l = *(const uint4*)g;       // load + ds_store fallback
#endif
}

// ---------------------------------------------------------------------------
// Kernel 1: fold sparse mask into W, convert W and x to bf16 in workspace.
// ---------------------------------------------------------------------------
__global__ void SP_prep_kernel(const float* __restrict__ x,
                               const float* __restrict__ W,
                               const float* __restrict__ mask,
                               unsigned short* __restrict__ xb,
                               unsigned short* __restrict__ wb) {
  size_t i      = (size_t)blockIdx.x * blockDim.x + threadIdx.x;
  size_t stride = (size_t)gridDim.x * blockDim.x;
  const size_t nx = (size_t)BATCH * DIN;
  const size_t nw = (size_t)DOUT * DIN;
  for (size_t j = i; j < nx; j += stride) xb[j] = f2bf(x[j]);
  for (size_t j = i; j < nw; j += stride) wb[j] = f2bf(W[j] * mask[j]);
}

// ---------------------------------------------------------------------------
// Kernel 2: y = xb @ wb^T + bias.
// Block tile 128(M) x 128(N), 256 threads = 8 wave32s as 4(M) x 2(N);
// each wave owns 32x64 (2x4 v_wmma_f32_16x16x32_bf16 accumulators).
// K-step 32: A tile 128x32 bf16 (8 KB) + B tile 128x32 bf16 (8 KB) are
// double-buffered in LDS via GLOBAL_LOAD_ASYNC_TO_LDS_B128 (ASYNCcnt),
// fragments read back as ds_load_b128 in the ISA 7.12.2 layouts:
//   A (16x32 bf16): lane<16 -> K {0..7,16..23}; lane>=16 -> K {8..15,24..31}
//   B (32x16 bf16): lane<16 -> K {0..7,16..23}; lane>=16 -> K {8..15,24..31}
//   C/D (16x16 f32): VGPR r, lane L: M = r + (L<16?0:8), N = L%16
// Staging loops are index-by-iteration (cc = it*256 + tid) so there are no
// per-lane bound checks -> straight-line async copies, no exec churn.
// ---------------------------------------------------------------------------
__device__ __forceinline__ void stage_tiles(const unsigned short* __restrict__ gA,
                                            const unsigned short* __restrict__ gB,
                                            unsigned short* lA,
                                            unsigned short* lB,
                                            int tid) {
  // 128 rows x 32 bf16 (64 B) per matrix = 512 16-byte chunks each;
  // 2 chunks per thread per matrix.
#pragma unroll
  for (int it = 0; it < 2; ++it) {
    const int cc  = it * 256 + tid;    // 0..511, no bound check needed
    const int row = cc >> 2;
    const int seg = cc & 3;
    copy16_to_lds(gA + (size_t)row * DIN + seg * 8, lA + row * 32 + seg * 8);
  }
#pragma unroll
  for (int it = 0; it < 2; ++it) {
    const int cc  = it * 256 + tid;
    const int row = cc >> 2;
    const int seg = cc & 3;
    copy16_to_lds(gB + (size_t)row * DIN + seg * 8, lB + row * 32 + seg * 8);
  }
}

__global__ __launch_bounds__(256)
void SP_gemm_kernel(const unsigned short* __restrict__ xb,
                    const unsigned short* __restrict__ wb,
                    const float* __restrict__ bias,
                    float* __restrict__ out) {
  __shared__ unsigned short Ab[2][128 * 32];   // 2 x 8 KB
  __shared__ unsigned short Bb[2][128 * 32];   // 2 x 8 KB

  const int tid   = threadIdx.x;
  const int lane  = tid & 31;
  const int wave  = tid >> 5;
  const int wm    = wave >> 1;                 // 0..3 -> M offset wm*32
  const int wn    = wave & 1;                  // 0..1 -> N offset wn*64
  const int l16   = lane & 15;
  const int hi    = lane >> 4;                 // 0 or 1

  const unsigned short* gA = xb + (size_t)(blockIdx.x * 128) * DIN;
  const unsigned short* gB = wb + (size_t)(blockIdx.y * 128) * DIN;

  v8f acc[2][4] = {};

  auto compute_step = [&](int cur) {
    v16bf afrag[2], bfrag[4];
#pragma unroll
    for (int mi = 0; mi < 2; ++mi) {
      const unsigned short* p = &Ab[cur][(wm * 32 + mi * 16 + l16) * 32 + hi * 8];
      v8bf lo = *(const v8bf*)(const void*)p;
      v8bf hb = *(const v8bf*)(const void*)(p + 16);
#pragma unroll
      for (int e = 0; e < 8; ++e) { afrag[mi][e] = lo[e]; afrag[mi][e + 8] = hb[e]; }
    }
#pragma unroll
    for (int ni = 0; ni < 4; ++ni) {
      const unsigned short* p = &Bb[cur][(wn * 64 + ni * 16 + l16) * 32 + hi * 8];
      v8bf lo = *(const v8bf*)(const void*)p;
      v8bf hb = *(const v8bf*)(const void*)(p + 16);
#pragma unroll
      for (int e = 0; e < 8; ++e) { bfrag[ni][e] = lo[e]; bfrag[ni][e + 8] = hb[e]; }
    }
#pragma unroll
    for (int mi = 0; mi < 2; ++mi)
#pragma unroll
      for (int ni = 0; ni < 4; ++ni)
        acc[mi][ni] = __builtin_amdgcn_wmma_f32_16x16x32_bf16(
            false, afrag[mi], false, bfrag[ni],
            (short)0, acc[mi][ni], false, false);
  };

  // Prologue: stage K-step 0 into buffer 0.
  stage_tiles(gA, gB, Ab[0], Bb[0], tid);
  wait_async_all();
  __syncthreads();

  // Steady state: stage next tile (async) while computing current one.
  for (int kk = 0; kk < DIN - 32; kk += 32) {
    const int cur = (kk >> 5) & 1;
    stage_tiles(gA + kk + 32, gB + kk + 32, Ab[cur ^ 1], Bb[cur ^ 1], tid);
    compute_step(cur);
    wait_async_all();       // next tile landed (issued before compute)
    __syncthreads();        // everyone done reading 'cur' before it is reused
  }
  // Peeled last K-step: no staging, no trailing sync needed.
  compute_step(((DIN / 32) - 1) & 1);

  // Epilogue: add bias, store f32 y.
#pragma unroll
  for (int mi = 0; mi < 2; ++mi) {
#pragma unroll
    for (int ni = 0; ni < 4; ++ni) {
      const int n  = blockIdx.y * 128 + wn * 64 + ni * 16 + l16;
      const float bv = bias[n];
#pragma unroll
      for (int r = 0; r < 8; ++r) {
        const int m = blockIdx.x * 128 + wm * 32 + mi * 16 + r + hi * 8;
        out[(size_t)m * DOUT + n] = acc[mi][ni][r] + bv;
      }
    }
  }
}

// ---------------------------------------------------------------------------
// Kernel 3: per-row k-winners. One 256-thread block per row.
// Map boosted f32 to order-preserving u32, 4-pass MSB radix select in LDS
// finds the exact k-th largest, then zero losers in place (winners keep the
// ORIGINAL y, matching the reference's where(boosted >= thresh, y, 0)).
// ---------------------------------------------------------------------------
__global__ __launch_bounds__(256)
void SP_kwin_kernel(float* __restrict__ out,
                    const float* __restrict__ duty,
                    const int* __restrict__ kptr) {
  __shared__ unsigned su[DOUT];        // 16 KB sortable keys
  __shared__ unsigned hist[256];
  __shared__ unsigned s_prefix, s_remaining;

  const int row = blockIdx.x;
  const int tid = threadIdx.x;
  const int k   = kptr[0];
  const float target = (float)k / (float)DOUT;
  float* yrow = out + (size_t)row * DOUT;

  for (int j = tid; j < DOUT; j += 256) {
    const float y   = yrow[j];
    const float bst = y * __expf(0.5f * (target - duty[j]));
    unsigned u = __float_as_uint(bst);
    u = (u & 0x80000000u) ? ~u : (u | 0x80000000u);   // order-preserving
    su[j] = u;
  }
  if (tid == 0) { s_prefix = 0u; s_remaining = (unsigned)k; }
  __syncthreads();

  unsigned prefmask = 0u;
  for (int pass = 3; pass >= 0; --pass) {
    const int shift = pass * 8;
    hist[tid] = 0u;
    __syncthreads();
    const unsigned pref = s_prefix;
    for (int j = tid; j < DOUT; j += 256) {
      const unsigned u = su[j];
      if ((u & prefmask) == pref)
        atomicAdd(&hist[(u >> shift) & 255u], 1u);
    }
    __syncthreads();
    if (tid == 0) {
      unsigned rem = s_remaining, cum = 0u; int sel = 0;
      for (int b = 255; b >= 0; --b) {
        if (cum + hist[b] >= rem) { sel = b; s_remaining = rem - cum; break; }
        cum += hist[b];
      }
      s_prefix = pref | ((unsigned)sel << shift);
    }
    __syncthreads();
    prefmask |= (0xFFu << shift);
  }

  const unsigned thresh = s_prefix;   // key of the k-th largest boosted value
  for (int j = tid; j < DOUT; j += 256) {
    if (su[j] < thresh) yrow[j] = 0.0f;   // winners keep original y
  }
}

// ---------------------------------------------------------------------------
extern "C" void kernel_launch(void* const* d_in, const int* in_sizes, int n_in,
                              void* d_out, int out_size, void* d_ws, size_t ws_size,
                              hipStream_t stream) {
  const float* x     = (const float*)d_in[0];
  const float* W     = (const float*)d_in[1];
  const float* b     = (const float*)d_in[2];
  const float* wmask = (const float*)d_in[3];
  const float* duty  = (const float*)d_in[4];
  const int*   kptr  = (const int*)d_in[5];
  float* out = (float*)d_out;

  // workspace: bf16 x (16 MB) then bf16 masked W (4 MB)
  unsigned short* xb = (unsigned short*)d_ws;
  unsigned short* wb = (unsigned short*)((char*)d_ws + (size_t)BATCH * DIN * 2);

  SP_prep_kernel<<<2048, 256, 0, stream>>>(x, W, wmask, xb, wb);

  dim3 gg(BATCH / 128, DOUT / 128);
  SP_gemm_kernel<<<gg, 256, 0, stream>>>(xb, wb, b, out);

  SP_kwin_kernel<<<BATCH, 256, 0, stream>>>(out, duty, kptr);
}
